// MaskedSelfAttention_42296837931429
// MI455X (gfx1250) — matrix operand order repaired and checked by reading, not compiled, not run on hardware
//
#include <hip/hip_runtime.h>
#include <hip/hip_bf16.h>
#include <math.h>

// ---------------------------------------------------------------------------
// MaskedSelfAttention for MI455X (gfx1250, wave32, WMMA bf16 w/ f32 accum)
//   S = 8192, D = 2048, inputs fp32. The reference's mask is a no-op.
//   Pipeline:
//     1) fp32 -> bf16 conversion of X, Wq, Wk, Wv
//     2) Q = X Wq^T, K = X Wk^T (bf16, row-major); V^T = (X Wv^T)^T (bf16)
//        - register-blocked WMMA GEMM, W panel staged by the Tensor Data
//          Mover (tensor_load_to_lds) into double-buffered LDS
//     3) flash attention: per 16-row query block, online softmax, O = P V
// ---------------------------------------------------------------------------

#define SEQ 8192
#define DM  2048

typedef __attribute__((ext_vector_type(16))) __bf16 bf16x16;
typedef __attribute__((ext_vector_type(8)))  __bf16 bf16x8;
typedef __attribute__((ext_vector_type(8)))  float  f32x8;
typedef __attribute__((ext_vector_type(4)))  unsigned int u32x4;
typedef __attribute__((ext_vector_type(8)))  int i32x8;
typedef __attribute__((ext_vector_type(4)))  int i32x4;

// A fragment (16x32 bf16, row-major source, leading dim ld):
// lane (m = lane&15, hf = lane>>4) holds row m, K = hf*8+0..7 and 16+hf*8+0..7.
__device__ __forceinline__ bf16x16
load_frag_a(const __bf16* __restrict__ p, int ld, int row0, int k0, int lane) {
  int m = lane & 15, hf = lane >> 4;
  const __bf16* base = p + (size_t)(row0 + m) * ld + k0 + hf * 8;
  union { bf16x16 v; bf16x8 h[2]; } u;
  u.h[0] = *(const bf16x8*)(base);
  u.h[1] = *(const bf16x8*)(base + 16);
  return u.v;
}

// B fragment (32x16 bf16). B[k][n] = Src[n0+n][k0+k]: columns of B are rows of
// the row-major source (W rows / K rows / V^T rows), contiguous along K.
// lane (c = lane&15, hf = lane>>4) holds column c, K = hf*8+0..7 and 16+hf*8+0..7.
__device__ __forceinline__ bf16x16
load_frag_b(const __bf16* __restrict__ p, int ld, int n0, int k0, int lane) {
  int c = lane & 15, hf = lane >> 4;
  const __bf16* base = p + (size_t)(n0 + c) * ld + k0 + hf * 8;
  union { bf16x16 v; bf16x8 h[2]; } u;
  u.h[0] = *(const bf16x8*)(base);
  u.h[1] = *(const bf16x8*)(base + 16);
  return u.v;
}

// B fragment from an LDS panel (row-major, ldl elements per row).
__device__ __forceinline__ bf16x16
load_frag_b_lds(const __bf16* p, int ldl, int n0, int k0, int lane) {
  int c = lane & 15, hf = lane >> 4;
  const __bf16* base = p + (n0 + c) * ldl + k0 + hf * 8;
  union { bf16x16 v; bf16x8 h[2]; } u;
  u.h[0] = *(const bf16x8*)(base);
  u.h[1] = *(const bf16x8*)(base + 16);
  return u.v;
}

__device__ __forceinline__ f32x8 wmma_bf16(bf16x16 a, bf16x16 b, f32x8 c) {
  return __builtin_amdgcn_wmma_f32_16x16x32_bf16(
      /*neg_a=*/false, a, /*neg_b=*/false, b,
      /*c_mod=*/(short)0, c, /*reuse_a=*/false, /*reuse_b=*/false);
}

// ---------------------------------------------------------------------------
// Tensor Data Mover: 2-D tile load (bf16 elements) per ISA D# layout.
//   group0: count=1 | lds_addr | global_addr[56:0] | type=2
//   group1: data_size=1(2B), tensor_dim0/1, tile_dim0/1, dim0 stride
//   groups 2/3 (+trailing group): zero (2-D tensor, tile_dim2..4 = 0)
// This toolchain exposes the 6-argument builtin form.
// ---------------------------------------------------------------------------
__device__ __forceinline__ void
tdm_load_2d_bf16(unsigned lds_byte_addr, const void* gptr,
                 unsigned tensor_d0, unsigned tensor_d1, unsigned stride_d0,
                 unsigned tile_d0, unsigned tile_d1) {
  unsigned long long ga = (unsigned long long)(uintptr_t)gptr;
  u32x4 g0;
  g0[0] = 0x1u;                                        // count=1 (valid)
  g0[1] = lds_byte_addr;                               // lds_addr
  g0[2] = (unsigned)(ga & 0xffffffffu);                // global_addr[31:0]
  g0[3] = (unsigned)((ga >> 32) & 0x01ffffffu) | (2u << 30);  // [56:32]|type=2
  i32x8 g1;
  g1[0] = (int)(1u << 16);                             // mask=0, data_size=2B
  g1[1] = (int)((tensor_d0 & 0xffffu) << 16);          // tensor_dim0 lo16
  g1[2] = (int)(((tensor_d0 >> 16) & 0xffffu) | ((tensor_d1 & 0xffffu) << 16));
  g1[3] = (int)(((tensor_d1 >> 16) & 0xffffu) | ((tile_d0 & 0xffffu) << 16));
  g1[4] = (int)(tile_d1 & 0xffffu);                    // tile_dim1, tile_dim2=0
  g1[5] = (int)stride_d0;                              // dim0 stride lo32
  g1[6] = 0;                                           // stride hi, dim1 stride
  g1[7] = 0;
  i32x4 gz4 = {0, 0, 0, 0};
  i32x8 gz8 = {0, 0, 0, 0, 0, 0, 0, 0};
  __builtin_amdgcn_tensor_load_to_lds(g0, g1, gz4, gz4, gz8, 0);
}

// ---------------------------------------------------------------------------
// Kernel 0: fp32 -> bf16 elementwise convert
// ---------------------------------------------------------------------------
__global__ void cvt_f32_to_bf16(const float* __restrict__ in,
                                __bf16* __restrict__ out, int n) {
  for (int i = blockIdx.x * blockDim.x + threadIdx.x; i < n;
       i += gridDim.x * blockDim.x)
    out[i] = (__bf16)in[i];
}

// ---------------------------------------------------------------------------
// Kernel 1: Out = X @ W^T (bf16 in, bf16 out).
// Block = 8 waves, macro tile 128(M) x 64(N). Wave w owns rows i0+16w..+15
// and all four 16-wide N tiles (A-frag reused 4x). The shared 64x64 W panel
// is staged into double-buffered LDS by the TDM (wave 0 issues, TENSORcnt
// + barrier synchronize). storeTransposed=1 writes Out^T (used for V).
// ---------------------------------------------------------------------------
__global__ __launch_bounds__(256)
void qkv_gemm(const __bf16* __restrict__ X, const __bf16* __restrict__ W,
              __bf16* __restrict__ Out, int storeTransposed) {
  __shared__ __bf16 wpanel[2][64 * 64];   // 2 x 8 KB double buffer

  const int lane = threadIdx.x & 31, wave = threadIdx.x >> 5;
  const int bn = blockIdx.x & 31;         // DM/64 = 32 N panels
  const int bm = blockIdx.x >> 5;         // SEQ/128 = 64 M panels
  const int i0 = bm * 128 + wave * 16;    // this wave's 16 rows
  const int n0 = bn * 64;                 // block's 64 output columns

  constexpr int KC = 64;                  // k chunk staged per TDM transfer
  constexpr int ITERS = DM / KC;          // 32

  // Prologue: stage first W panel (rows n0..n0+63, cols 0..63).
  if (wave == 0) {
    tdm_load_2d_bf16((unsigned)(uintptr_t)(void*)&wpanel[0][0],
                     W + (size_t)n0 * DM, DM, DM, DM, KC, 64);
  }

  f32x8 acc[4];
  f32x8 zero = {};
  for (int t = 0; t < 4; ++t) acc[t] = zero;

  for (int it = 0; it < ITERS; ++it) {
    const int cur = it & 1;
    if (it + 1 < ITERS) {
      if (wave == 0) {
        tdm_load_2d_bf16((unsigned)(uintptr_t)(void*)&wpanel[cur ^ 1][0],
                         W + (size_t)n0 * DM + (it + 1) * KC,
                         DM, DM, DM, KC, 64);
        __builtin_amdgcn_s_wait_tensorcnt(1);  // current panel complete
      }
    } else {
      if (wave == 0) __builtin_amdgcn_s_wait_tensorcnt(0);
    }
    __syncthreads();                            // panel visible to all waves

    const int kc = it * KC;
    for (int ks = 0; ks < KC; ks += 32) {
      bf16x16 a = load_frag_a(X, DM, i0, kc + ks, lane);
      for (int nt = 0; nt < 4; ++nt) {
        bf16x16 b = load_frag_b_lds(&wpanel[cur][0], KC, nt * 16, ks, lane);
        acc[nt] = wmma_bf16(a, b, acc[nt]);
      }
    }
    __syncthreads();                            // reads done before reuse
  }

  const int cn = lane & 15;
  const int rbase = i0 + ((lane >> 4) << 3);    // row = rbase + r
  for (int nt = 0; nt < 4; ++nt) {
    int col = n0 + nt * 16 + cn;
    if (!storeTransposed) {
      for (int r = 0; r < 8; ++r)
        Out[(size_t)(rbase + r) * DM + col] = (__bf16)acc[nt][r];
    } else {                                    // Out is [DM][SEQ]
      for (int r = 0; r < 8; ++r)
        Out[(size_t)col * SEQ + (rbase + r)] = (__bf16)acc[nt][r];
    }
  }
}

// ---------------------------------------------------------------------------
// Kernel 2: flash attention. Block = 8 waves = one 16-row query block.
//  * Q.K^T: 2048-deep dot split across the 8 waves (256 each); partial 16x32
//    score tiles summed through LDS.
//  * online softmax per row (running max/sum in registers, cross-lane
//    reductions via __shfl_xor over 16-lane half-groups).
//  * P.V: each wave owns a 256-wide slice of O; P repacked fp32->bf16 through
//    a per-wave LDS tile into the A-fragment layout.
// ---------------------------------------------------------------------------
__global__ __launch_bounds__(256, 1)
void flash_attn(const __bf16* __restrict__ Q, const __bf16* __restrict__ K,
                const __bf16* __restrict__ Vt, float* __restrict__ O) {
  __shared__ float  sbuf[8 * 2 * 32 * 8];   // 16 KB: partial score tiles
  __shared__ __bf16 pbuf[8][16][32];        //  8 KB: per-wave P staging

  const int lane = threadIdx.x & 31, wave = threadIdx.x >> 5;
  const int hf = lane >> 4, cn = lane & 15;
  const int i0 = blockIdx.x * 16;           // query rows [i0, i0+16)
  const int kbase = wave * 256;             // this wave's slice of the QK dot
  const int d0 = wave * 256;                // this wave's output column slice
  const float scale = 0.022097086912079608f; // 1/sqrt(2048)

  bf16x16 qf[8];
  for (int kk = 0; kk < 8; ++kk)
    qf[kk] = load_frag_a(Q, DM, i0, kbase + kk * 32, lane);

  f32x8 zero = {};
  f32x8 oacc[16];
  for (int t = 0; t < 16; ++t) oacc[t] = zero;
  float mrow[8], lrow[8];
  for (int r = 0; r < 8; ++r) { mrow[r] = -INFINITY; lrow[r] = 0.0f; }

  for (int j0 = 0; j0 < SEQ; j0 += 32) {
    // ---- partial scores for key columns [j0, j0+32) over our K-slice ----
    f32x8 s0 = zero, s1 = zero;
    for (int kk = 0; kk < 8; ++kk) {
      bf16x16 b0 = load_frag_b(K, DM, j0,      kbase + kk * 32, lane);
      bf16x16 b1 = load_frag_b(K, DM, j0 + 16, kbase + kk * 32, lane);
      s0 = wmma_bf16(qf[kk], b0, s0);
      s1 = wmma_bf16(qf[kk], b1, s1);
    }
    if (j0 + 32 < SEQ) {  // hint next K/V tiles toward the caches
      __builtin_prefetch(K + (size_t)(j0 + 32 + cn) * DM + kbase + hf * 512, 0, 1);
      __builtin_prefetch(Vt + (size_t)(d0 + cn) * SEQ + j0 + 32, 0, 1);
    }

    *(f32x8*)&sbuf[((wave * 2 + 0) * 32 + lane) * 8] = s0;
    *(f32x8*)&sbuf[((wave * 2 + 1) * 32 + lane) * 8] = s1;
    __syncthreads();

    f32x8 S0 = zero, S1 = zero;
    for (int ww = 0; ww < 8; ++ww) {
      S0 += *(const f32x8*)&sbuf[((ww * 2 + 0) * 32 + lane) * 8];
      S1 += *(const f32x8*)&sbuf[((ww * 2 + 1) * 32 + lane) * 8];
    }

    // ---- online softmax over these 32 columns ----
    float alpha[8];
    f32x8 P0, P1;
    for (int r = 0; r < 8; ++r) {
      float a0 = S0[r] * scale, a1 = S1[r] * scale;
      float mx = fmaxf(a0, a1);
      mx = fmaxf(mx, __shfl_xor(mx, 8, 32));
      mx = fmaxf(mx, __shfl_xor(mx, 4, 32));
      mx = fmaxf(mx, __shfl_xor(mx, 2, 32));
      mx = fmaxf(mx, __shfl_xor(mx, 1, 32));
      float mnew = fmaxf(mrow[r], mx);
      float p0 = __expf(a0 - mnew);
      float p1 = __expf(a1 - mnew);
      float ps = p0 + p1;
      ps += __shfl_xor(ps, 8, 32);
      ps += __shfl_xor(ps, 4, 32);
      ps += __shfl_xor(ps, 2, 32);
      ps += __shfl_xor(ps, 1, 32);
      alpha[r] = __expf(mrow[r] - mnew);
      lrow[r] = lrow[r] * alpha[r] + ps;
      mrow[r] = mnew;
      P0[r] = p0;
      P1[r] = p1;
    }
    for (int t = 0; t < 16; ++t)
      for (int r = 0; r < 8; ++r) oacc[t][r] *= alpha[r];

    // ---- stage P (16x32) to LDS in row-major bf16 ----
    for (int r = 0; r < 8; ++r) {
      pbuf[wave][hf * 8 + r][cn]      = (__bf16)P0[r];
      pbuf[wave][hf * 8 + r][16 + cn] = (__bf16)P1[r];
    }
    __syncthreads();  // orders pbuf store->load and sbuf read->next write

    bf16x16 pf;
    {
      const __bf16* base = &pbuf[wave][cn][hf * 8];
      union { bf16x16 v; bf16x8 h[2]; } u;
      u.h[0] = *(const bf16x8*)(base);
      u.h[1] = *(const bf16x8*)(base + 16);
      pf = u.v;
    }

    // ---- O slice += P * V : B columns are V^T rows (contiguous in j) ----
    for (int t = 0; t < 16; ++t) {
      bf16x16 bv = load_frag_b(Vt, SEQ, d0 + t * 16, j0, lane);
      oacc[t] = wmma_bf16(pf, bv, oacc[t]);
    }
  }

  float rinv[8];
  for (int r = 0; r < 8; ++r) rinv[r] = 1.0f / lrow[r];
  int rbase = i0 + hf * 8;
  for (int t = 0; t < 16; ++t) {
    int col = d0 + t * 16 + cn;
    for (int r = 0; r < 8; ++r)
      O[(size_t)(rbase + r) * DM + col] = oacc[t][r] * rinv[r];
  }
}

// ---------------------------------------------------------------------------
extern "C" void kernel_launch(void* const* d_in, const int* in_sizes, int n_in,
                              void* d_out, int out_size, void* d_ws, size_t ws_size,
                              hipStream_t stream) {
  const float* X  = (const float*)d_in[0];  // [SEQ, DM]
  const float* Wq = (const float*)d_in[1];  // [DM, DM]
  const float* Wk = (const float*)d_in[2];
  const float* Wv = (const float*)d_in[3];
  float* Out = (float*)d_out;               // [SEQ, DM]

  char* ws = (char*)d_ws;
  const size_t XD = (size_t)SEQ * DM;       // 16,777,216 elems
  const size_t WD = (size_t)DM * DM;        //  4,194,304 elems
  __bf16* Xbf  = (__bf16*)(ws);
  __bf16* Wqbf = (__bf16*)(ws + 2 * XD);
  __bf16* Wkbf = (__bf16*)(ws + 2 * XD + 2 * WD);
  __bf16* Wvbf = (__bf16*)(ws + 2 * XD + 4 * WD);
  __bf16* Qbf  = (__bf16*)(ws + 2 * XD + 6 * WD);
  __bf16* Kbf  = (__bf16*)(ws + 4 * XD + 6 * WD);
  __bf16* Vt   = (__bf16*)(ws + 6 * XD + 6 * WD);  // stored [DM][SEQ]

  cvt_f32_to_bf16<<<2048, 256, 0, stream>>>(X,  Xbf,  (int)XD);
  cvt_f32_to_bf16<<<1024, 256, 0, stream>>>(Wq, Wqbf, (int)WD);
  cvt_f32_to_bf16<<<1024, 256, 0, stream>>>(Wk, Wkbf, (int)WD);
  cvt_f32_to_bf16<<<1024, 256, 0, stream>>>(Wv, Wvbf, (int)WD);

  // (SEQ/128)*(DM/64) = 64*32 = 2048 blocks of 8 waves.
  qkv_gemm<<<2048, 256, 0, stream>>>(Xbf, Wqbf, Qbf, 0);
  qkv_gemm<<<2048, 256, 0, stream>>>(Xbf, Wkbf, Kbf, 0);
  qkv_gemm<<<2048, 256, 0, stream>>>(Xbf, Wvbf, Vt, 1);

  flash_attn<<<SEQ / 16, 256, 0, stream>>>(Qbf, Kbf, Vt, Out);
}
